// Net_67430986547796
// MI455X (gfx1250) — matrix operand order repaired and checked by reading, not compiled, not run on hardware
//
#include <hip/hip_runtime.h>
#include <hip/hip_bf16.h>

typedef __attribute__((ext_vector_type(16))) _Float16 v16h;
typedef __attribute__((ext_vector_type(8)))  float    v8f;

#define NFEAT 512
#define NHID  16
#define NCLS  10

// ---------------- degree init ----------------
__global__ void init_deg_kernel(float* deg, int n) {
  int i = blockIdx.x * blockDim.x + threadIdx.x;
  if (i < n) deg[i] = 1.0f;               // self-loop contributes 1 to in-degree
}

// ---------------- fused: int64 edges -> interleaved int2, + in-degree count ----------------
__global__ void convert_count_kernel(const long long* __restrict__ ei64,
                                     int2* __restrict__ ec, float* deg, int E) {
  int e = blockIdx.x * blockDim.x + threadIdx.x;
  if (e >= E) return;
  int r = (int)ei64[e];
  int c = (int)ei64[(size_t)E + e];
  ec[e] = make_int2(r, c);
  atomicAdd(&deg[c], 1.0f);
}

__global__ void rsqrt_deg_kernel(float* deg, int n) {
  int i = blockIdx.x * blockDim.x + threadIdx.x;
  if (i < n) {
    float d = deg[i];
    deg[i] = (d > 0.0f) ? rsqrtf(d) : 0.0f;   // now holds dinv
  }
}

// ---------------- GEMM1: h1 = x @ W1  (100000x512 @ 512x16), WMMA f16->f32 ----------------
// One wave computes a 16x16 tile. K loop: 16 chunks of 32.
// W1 is staged in LDS pre-swizzled into WMMA B-fragment order so each lane
// reads its whole fragment as one contiguous 32B (2x ds_load_b128).
__global__ void __launch_bounds__(256) gemm1_wmma_kernel(
    const float* __restrict__ x, const float* __restrict__ W1,
    float* __restrict__ h1, int n_tiles)
{
  __shared__ _Float16 sB[16 * 32 * 16];   // [chunk][lane][e] = 16 KB
  for (int i = threadIdx.x; i < 16 * 32 * 16; i += 256) {
    int c   = i >> 9;          // chunk
    int rem = i & 511;
    int l   = rem >> 4;        // lane
    int e   = rem & 15;        // element within fragment
    int hi  = l >> 4;
    int nb  = l & 15;          // column N owned by lane l
    // B element e of lane l holds W1[K = c*32 + hi*8 + (e&7) + (e>>3)*16][N = nb]
    sB[i] = (_Float16)W1[(c * 32 + hi * 8 + (e & 7) + (e >> 3) * 16) * NHID + nb];
  }
  __syncthreads();

  const int wave = threadIdx.x >> 5;
  const int lane = threadIdx.x & 31;
  const int tile = blockIdx.x * 8 + wave;
  if (tile >= n_tiles) return;

  const int m  = lane & 15;     // A row this lane owns
  const int hi = lane >> 4;     // lane-half selects K sub-ranges
  const float* __restrict__ xrow = x + (size_t)(tile * 16 + m) * NFEAT;
  const _Float16* __restrict__ myB = sB + lane * 16;

  v8f acc = {};
  #pragma unroll 4
  for (int c = 0; c < 16; ++c) {
    // A fragment (16x32 f16): elems 0..7 -> K = c*32 + hi*8 + e ; elems 8..15 -> +16
    const int k0 = c * 32 + hi * 8;
    float4 a0 = *reinterpret_cast<const float4*>(xrow + k0);
    float4 a1 = *reinterpret_cast<const float4*>(xrow + k0 + 4);
    float4 a2 = *reinterpret_cast<const float4*>(xrow + k0 + 16);
    float4 a3 = *reinterpret_cast<const float4*>(xrow + k0 + 20);
    v16h A;
    A[0]=(_Float16)a0.x; A[1]=(_Float16)a0.y; A[2]=(_Float16)a0.z; A[3]=(_Float16)a0.w;
    A[4]=(_Float16)a1.x; A[5]=(_Float16)a1.y; A[6]=(_Float16)a1.z; A[7]=(_Float16)a1.w;
    A[8]=(_Float16)a2.x; A[9]=(_Float16)a2.y; A[10]=(_Float16)a2.z; A[11]=(_Float16)a2.w;
    A[12]=(_Float16)a3.x; A[13]=(_Float16)a3.y; A[14]=(_Float16)a3.z; A[15]=(_Float16)a3.w;

    // whole B fragment for this lane: contiguous 32 bytes in LDS
    v16h B = *reinterpret_cast<const v16h*>(myB + c * 32 * 16);

    acc = __builtin_amdgcn_wmma_f32_16x16x32_f16(false, A, false, B,
                                                 (short)0, acc, false, false);
  }

  // D layout: VGPR r, lanes 0-15: M=r N=lane ; lanes 16-31: M=r+8 N=lane-16
  float* __restrict__ out = h1 + (size_t)tile * 16 * NHID;
  #pragma unroll
  for (int r = 0; r < 8; ++r)
    out[(r + hi * 8) * NHID + (lane & 15)] = acc[r];
}

// ---------------- layer-1 propagate ----------------
// out1[i][f] = h1[i][f]*dinv[i]^2 + b1[f]   (self loop + bias), then edge atomics.
__global__ void scatter1_init_kernel(const float* __restrict__ h1,
                                     const float* __restrict__ dinv,
                                     const float* __restrict__ b1,
                                     float* __restrict__ out1, int n)
{
  int t = blockIdx.x * blockDim.x + threadIdx.x;
  if (t >= n * NHID) return;
  int i = t >> 4, f = t & 15;
  float di = dinv[i];
  out1[t] = h1[t] * di * di + b1[f];
}

__global__ void scatter1_edges_kernel(const int2* __restrict__ ec,
                                      const float* __restrict__ h1,
                                      const float* __restrict__ dinv,
                                      float* __restrict__ out1, int E)
{
  int e = blockIdx.x * blockDim.x + threadIdx.x;
  if (e >= E) return;
  int2 rc = ec[e];                        // single b64 load per edge
  float norm = dinv[rc.x] * dinv[rc.y];
  const float4* __restrict__ hr = reinterpret_cast<const float4*>(h1 + (size_t)rc.x * NHID);
  float4 v0 = hr[0], v1 = hr[1], v2 = hr[2], v3 = hr[3];
  float* __restrict__ dst = out1 + (size_t)rc.y * NHID;
  atomicAdd(dst + 0,  v0.x * norm); atomicAdd(dst + 1,  v0.y * norm);
  atomicAdd(dst + 2,  v0.z * norm); atomicAdd(dst + 3,  v0.w * norm);
  atomicAdd(dst + 4,  v1.x * norm); atomicAdd(dst + 5,  v1.y * norm);
  atomicAdd(dst + 6,  v1.z * norm); atomicAdd(dst + 7,  v1.w * norm);
  atomicAdd(dst + 8,  v2.x * norm); atomicAdd(dst + 9,  v2.y * norm);
  atomicAdd(dst + 10, v2.z * norm); atomicAdd(dst + 11, v2.w * norm);
  atomicAdd(dst + 12, v3.x * norm); atomicAdd(dst + 13, v3.y * norm);
  atomicAdd(dst + 14, v3.z * norm); atomicAdd(dst + 15, v3.w * norm);
}

// ---------------- GEMM2: h2 = relu(out1) @ W2  (K=16 padded to 32, N=10 padded to 16) ----------------
__global__ void __launch_bounds__(256) gemm2_wmma_kernel(
    const float* __restrict__ out1, const float* __restrict__ W2,
    float* __restrict__ h2, int n_tiles)
{
  __shared__ _Float16 sB[32 * 16];        // fragment-order padded W2: [lane][e]
  for (int i = threadIdx.x; i < 32 * 16; i += 256) {
    int l  = i >> 4;           // lane
    int e  = i & 15;           // element
    int hi = l >> 4;
    int nb = l & 15;
    // B element e of lane l holds W2pad[K = hi*8 + (e&7) + (e>>3)*16][N = nb]; real only K<16 (e<8), N<10
    float v = (e < 8 && nb < NCLS) ? W2[(hi * 8 + e) * NCLS + nb] : 0.0f;
    sB[i] = (_Float16)v;
  }
  __syncthreads();

  const int wave = threadIdx.x >> 5;
  const int lane = threadIdx.x & 31;
  const int tile = blockIdx.x * 8 + wave;
  if (tile >= n_tiles) return;

  const int m  = lane & 15;
  const int hi = lane >> 4;
  const float* __restrict__ arow = out1 + (size_t)(tile * 16 + m) * NHID;

  // A: elems 0..7 -> K = hi*8 + e (< 16, fused ReLU); elems 8..15 -> K >= 16, zero pad
  float4 a0 = *reinterpret_cast<const float4*>(arow + hi * 8);
  float4 a1 = *reinterpret_cast<const float4*>(arow + hi * 8 + 4);
  v16h A;
  A[0]=(_Float16)fmaxf(a0.x,0.f); A[1]=(_Float16)fmaxf(a0.y,0.f);
  A[2]=(_Float16)fmaxf(a0.z,0.f); A[3]=(_Float16)fmaxf(a0.w,0.f);
  A[4]=(_Float16)fmaxf(a1.x,0.f); A[5]=(_Float16)fmaxf(a1.y,0.f);
  A[6]=(_Float16)fmaxf(a1.z,0.f); A[7]=(_Float16)fmaxf(a1.w,0.f);
  #pragma unroll
  for (int e = 8; e < 16; ++e) A[e] = (_Float16)0.0f;

  v16h B = *reinterpret_cast<const v16h*>(sB + lane * 16);

  v8f acc = {};
  acc = __builtin_amdgcn_wmma_f32_16x16x32_f16(false, A, false, B,
                                               (short)0, acc, false, false);

  float* __restrict__ out = h2 + (size_t)tile * 16 * NHID;   // stride-16 rows, cols 10..15 zero
  #pragma unroll
  for (int r = 0; r < 8; ++r)
    out[(r + hi * 8) * NHID + (lane & 15)] = acc[r];
}

// ---------------- layer-2 propagate (writes into d_out, stride NCLS) ----------------
__global__ void scatter2_init_kernel(const float* __restrict__ h2,
                                     const float* __restrict__ dinv,
                                     const float* __restrict__ b2,
                                     float* __restrict__ out2, int n)
{
  int t = blockIdx.x * blockDim.x + threadIdx.x;
  if (t >= n * NCLS) return;
  int i = t / NCLS, c = t % NCLS;
  float di = dinv[i];
  out2[t] = h2[(size_t)i * NHID + c] * di * di + b2[c];
}

__global__ void scatter2_edges_kernel(const int2* __restrict__ ec,
                                      const float* __restrict__ h2,
                                      const float* __restrict__ dinv,
                                      float* __restrict__ out2, int E)
{
  int e = blockIdx.x * blockDim.x + threadIdx.x;
  if (e >= E) return;
  int2 rc = ec[e];
  float norm = dinv[rc.x] * dinv[rc.y];
  const float* __restrict__ hr = h2 + (size_t)rc.x * NHID;
  float4 v0 = *reinterpret_cast<const float4*>(hr);
  float4 v1 = *reinterpret_cast<const float4*>(hr + 4);
  float2 v2 = *reinterpret_cast<const float2*>(hr + 8);
  float* __restrict__ dst = out2 + (size_t)rc.y * NCLS;
  atomicAdd(dst + 0, v0.x * norm); atomicAdd(dst + 1, v0.y * norm);
  atomicAdd(dst + 2, v0.z * norm); atomicAdd(dst + 3, v0.w * norm);
  atomicAdd(dst + 4, v1.x * norm); atomicAdd(dst + 5, v1.y * norm);
  atomicAdd(dst + 6, v1.z * norm); atomicAdd(dst + 7, v1.w * norm);
  atomicAdd(dst + 8, v2.x * norm); atomicAdd(dst + 9, v2.y * norm);
}

// ---------------- log_softmax over 10 classes, in place on d_out ----------------
__global__ void logsoftmax_kernel(float* __restrict__ out, int n)
{
  int i = blockIdx.x * blockDim.x + threadIdx.x;
  if (i >= n) return;
  float* row = out + (size_t)i * NCLS;
  float v[NCLS];
  float m = -__builtin_inff();
  #pragma unroll
  for (int c = 0; c < NCLS; ++c) { v[c] = row[c]; m = fmaxf(m, v[c]); }
  float s = 0.0f;
  #pragma unroll
  for (int c = 0; c < NCLS; ++c) s += __expf(v[c] - m);
  float l = __logf(s);
  #pragma unroll
  for (int c = 0; c < NCLS; ++c) row[c] = v[c] - m - l;
}

// ---------------- launch ----------------
extern "C" void kernel_launch(void* const* d_in, const int* in_sizes, int n_in,
                              void* d_out, int out_size, void* d_ws, size_t ws_size,
                              hipStream_t stream) {
  const float*     x    = (const float*)d_in[0];
  const long long* ei64 = (const long long*)d_in[1];
  const float*     W1   = (const float*)d_in[2];
  const float*     b1   = (const float*)d_in[3];
  const float*     W2   = (const float*)d_in[4];
  const float*     b2   = (const float*)d_in[5];
  float* out = (float*)d_out;

  const int N = in_sizes[0] / NFEAT;    // 100000
  const int E = in_sizes[1] / 2;        // 3200000
  const int n_tiles = (N + 15) / 16;    // 6250

  // workspace carve-up (256B aligned)
  char* ws = (char*)d_ws;
  size_t o = 0;
  float* dinv = (float*)(ws + o); o += ((size_t)N * 4 + 255) & ~(size_t)255;
  float* h1   = (float*)(ws + o); o += ((size_t)N * NHID * 4 + 255) & ~(size_t)255;
  float* out1 = (float*)(ws + o); o += ((size_t)N * NHID * 4 + 255) & ~(size_t)255;
  float* h2   = (float*)(ws + o); o += ((size_t)N * NHID * 4 + 255) & ~(size_t)255;
  int2*  ec   = (int2*)(ws + o);  o += ((size_t)E * 8 + 255) & ~(size_t)255;
  (void)ws_size; (void)n_in; (void)out_size;

  const int B = 256;
  // degrees -> dinv, edge list -> int2 (fused)
  init_deg_kernel     <<<(N + B - 1) / B, B, 0, stream>>>(dinv, N);
  convert_count_kernel<<<(E + B - 1) / B, B, 0, stream>>>(ei64, ec, dinv, E);
  rsqrt_deg_kernel    <<<(N + B - 1) / B, B, 0, stream>>>(dinv, N);

  // layer 1
  gemm1_wmma_kernel<<<(n_tiles + 7) / 8, B, 0, stream>>>(x, W1, h1, n_tiles);
  scatter1_init_kernel <<<((size_t)N * NHID + B - 1) / B, B, 0, stream>>>(h1, dinv, b1, out1, N);
  scatter1_edges_kernel<<<(E + B - 1) / B, B, 0, stream>>>(ec, h1, dinv, out1, E);

  // layer 2 (ReLU fused into GEMM2 A-load)
  gemm2_wmma_kernel<<<(n_tiles + 7) / 8, B, 0, stream>>>(out1, W2, h2, n_tiles);
  scatter2_init_kernel <<<((size_t)N * NCLS + B - 1) / B, B, 0, stream>>>(h2, dinv, b2, out, N);
  scatter2_edges_kernel<<<(E + B - 1) / B, B, 0, stream>>>(ec, h2, dinv, out, E);

  logsoftmax_kernel<<<(N + B - 1) / B, B, 0, stream>>>(out, N);
}
